// GraphAttnBias_65420941853217
// MI455X (gfx1250) — matrix operand order repaired and checked by reading, not compiled, not run on hardware
//
#include <hip/hip_runtime.h>
#include <hip/hip_bf16.h>

// ---------------------------------------------------------------------------
// Graphormer attention-bias kernel for gfx1250 (MI455X).
// Interior edge-encoding matmul mapped to v_wmma_f32_16x16x32_bf16.
// ---------------------------------------------------------------------------

typedef __attribute__((ext_vector_type(16))) __bf16 v16bf;
typedef __attribute__((ext_vector_type(8)))  float  v8f;
typedef __attribute__((ext_vector_type(4)))  float  v4f;

#define GB_B    32
#define GB_N    128
#define GB_H    32
#define GB_NP1  129
#define GB_SQ   (GB_NP1 * GB_NP1)     // 16641
#define GB_MHMD 5

#define WAVES_PER_BLOCK 8
#define TILES_PER_WAVE  8

// ---------------------------------------------------------------------------
// Interior kernel: one wave32 handles a tile of 16 consecutive j-positions.
//   edge[m, :] = ( sum_d  A_d[m,:] (16x32 bf16)  x  (Wd[d]/3) ) / spc[m]
//   out[b,h,i+1,j+1] = 2*ab + spatial_bias + edge
// (the 1/3 feature-mean is folded into the B fragments, built once per wave)
// ---------------------------------------------------------------------------
__global__ __launch_bounds__(256) void gb_interior_kernel(
    const float* __restrict__ ab,     // [B,129,129]
    const int*   __restrict__ spm,    // [B,128,128]
    const int*   __restrict__ eidx,   // [B,128,128,5,3]
    const float* __restrict__ Ew,     // [1537,32]
    const float* __restrict__ Wd,     // [128,32,32] flat
    const float* __restrict__ Sw,     // [512,32]
    float*       __restrict__ out)    // [B,32,129,129]
{
  __shared__ float lds[WAVES_PER_BLOCK][16 * 34];   // padded: stride 34 -> no bank conflicts

  const int lane = threadIdx.x & 31;
  const int wave = threadIdx.x >> 5;
  const int half = lane >> 4;       // 0: lanes 0-15, 1: lanes 16-31
  const int l16  = lane & 15;

  // ---- Preload B fragments (Wd/3, converted to bf16) once per wave --------
  // B layout (32x16, 16-bit): half0 elems -> K {0..7,16..23},
  // half1 elems -> K {8..15,24..31} ; N = lane%16.   B[k][n] = Wd[d*1024 + k*32 + n]
  v16bf Bfrag[GB_MHMD][2];
  for (int d = 0; d < GB_MHMD; ++d) {
    for (int nh = 0; nh < 2; ++nh) {
      union { v16bf v; __bf16 e[16]; } u;
      const int n = nh * 16 + l16;
      #pragma unroll
      for (int v = 0; v < 8; ++v) {
        const int h0 = 2 * v + half * 8 + ((v >= 4) ? 8 : 0);
        u.e[2 * v]     = (__bf16)(Wd[d * 1024 + h0 * 32 + n] * (1.0f / 3.0f));
        u.e[2 * v + 1] = (__bf16)(Wd[d * 1024 + (h0 + 1) * 32 + n] * (1.0f / 3.0f));
      }
      Bfrag[d][nh] = u.v;
    }
  }

  const int tbase = (blockIdx.x * WAVES_PER_BLOCK + wave) * TILES_PER_WAVE;

  for (int it = 0; it < TILES_PER_WAVE; ++it) {
    const int t  = tbase + it;
    const int m0 = t << 4;                       // first position of the tile
    const int j0 = m0 & (GB_N - 1);
    const int i  = (m0 >> 7) & (GB_N - 1);
    const int b  = m0 >> 14;

    const int m    = m0 + l16;                   // this lane's A-row position
    const int ibas = m * (GB_MHMD * 3);

    v8f acc0 = {};
    v8f acc1 = {};

    // ---- 5 distances: build A fragment, issue 2 WMMAs each ----------------
    #pragma unroll
    for (int d = 0; d < GB_MHMD; ++d) {
      float s[16];
      #pragma unroll
      for (int e = 0; e < 16; ++e) s[e] = 0.0f;

      #pragma unroll
      for (int f = 0; f < 3; ++f) {
        const int idx = eidx[ibas + d * 3 + f];          // 0..1536
        const float* row = Ew + idx * GB_H + half * 8;   // 16B-aligned chunks
        #pragma unroll
        for (int c = 0; c < 4; ++c) {
          const int kb = (c < 2) ? (c * 4) : (c * 4 + 8);   // 0,4,16,20 (+half*8)
          const v4f q = *(const v4f*)(row + kb);
          s[c * 4 + 0] += q.x;
          s[c * 4 + 1] += q.y;
          s[c * 4 + 2] += q.z;
          s[c * 4 + 3] += q.w;
        }
      }

      // A layout (16x32, 16-bit): half0 elems -> K {0..7,16..23},
      //                            half1 elems -> K {8..15,24..31}
      union { v16bf v; __bf16 e[16]; } ua;
      #pragma unroll
      for (int e = 0; e < 16; ++e) ua.e[e] = (__bf16)s[e];

      acc0 = __builtin_amdgcn_wmma_f32_16x16x32_bf16(
          false, ua.v, false, Bfrag[d][0], (short)0, acc0, false, false);
      acc1 = __builtin_amdgcn_wmma_f32_16x16x32_bf16(
          false, ua.v, false, Bfrag[d][1], (short)0, acc1, false, false);
    }

    // ---- transpose 16x32 accumulator tile through LDS ----------------------
    // C layout: VGPR r -> position p = r + 8*half ; head h = l16 (+16 for acc1)
    float* tp = lds[wave];
    #pragma unroll
    for (int r = 0; r < 8; ++r) {
      const int p = r + half * 8;
      tp[p * 34 + l16]      = acc0[r];
      tp[p * 34 + 16 + l16] = acc1[r];
    }
    asm volatile("s_wait_dscnt 0" ::: "memory");   // cross-lane LDS RAW

    // ---- per-lane j-column data -------------------------------------------
    const int   jj  = j0 + l16;
    const float abv = ab[b * GB_SQ + (i + 1) * GB_NP1 + (jj + 1)];
    const int   spv = spm[(b * GB_N + i) * GB_N + jj];
    int spc = (spv == 0) ? 1 : spv;
    spc = (spc > 1) ? (spc - 1) : spc;
    spc = (spc > GB_MHMD) ? GB_MHMD : spc;
    const float rsp   = 1.0f / (float)spc;
    const float base2 = 2.0f * abv;

    const long obase = (long)b * GB_H * GB_SQ + (long)(i + 1) * GB_NP1 + (jj + 1);

    // 16 passes; each half-wave stores 16 consecutive floats (coalesced 64B)
    #pragma unroll
    for (int hp = 0; hp < 16; ++hp) {
      const int   h    = 2 * hp + half;
      const float edge = tp[l16 * 34 + h] * rsp;
      const float spb  = Sw[spv * GB_H + h];
      out[obase + (long)h * GB_SQ] = base2 + spb + edge;
    }
  }
}

// ---------------------------------------------------------------------------
// Boundary kernel: row 0 (all 129 j) and column 0 (i = 1..128):
//   out = 2*attn_bias + graph_token_virtual_distance[h]
// ---------------------------------------------------------------------------
__global__ void gb_boundary_kernel(const float* __restrict__ ab,
                                   const float* __restrict__ gt,
                                   float* __restrict__ out)
{
  const int idx = blockIdx.x * blockDim.x + threadIdx.x;
  const int TOT = GB_B * GB_H * (GB_NP1 + GB_N);   // 32*32*257
  if (idx >= TOT) return;
  const int k = idx % 257;
  const int h = (idx / 257) % GB_H;
  const int b = idx / (257 * GB_H);
  int i, j;
  if (k < GB_NP1) { i = 0; j = k; }
  else            { i = k - (GB_NP1 - 1); j = 0; }
  const float v = 2.0f * ab[b * GB_SQ + i * GB_NP1 + j] + gt[h];
  out[((long)(b * GB_H + h) * GB_NP1 + i) * GB_NP1 + j] = v;
}

// ---------------------------------------------------------------------------
extern "C" void kernel_launch(void* const* d_in, const int* in_sizes, int n_in,
                              void* d_out, int out_size, void* d_ws, size_t ws_size,
                              hipStream_t stream) {
  (void)in_sizes; (void)n_in; (void)out_size; (void)d_ws; (void)ws_size;

  const float* ab   = (const float*)d_in[0];   // attn_bias            [32,129,129]
  const int*   spm  = (const int*)  d_in[1];   // spatial_pos          [32,128,128]
  // d_in[2] = x (unused by reference)
  const int*   eidx = (const int*)  d_in[3];   // edge_input           [32,128,128,5,3]
  // d_in[4] = attn_edge_type (unused by reference)
  const float* Ew   = (const float*)d_in[5];   // edge_encoder_w       [1537,32]
  const float* Wd   = (const float*)d_in[6];   // edge_dis_encoder_w   [131072]
  const float* Sw   = (const float*)d_in[7];   // spatial_pos_encoder_w[512,32]
  const float* gt   = (const float*)d_in[8];   // graph_token_virtual_distance_w [32]
  float* out = (float*)d_out;                  // [32,32,129,129]

  // interior: 32768 tiles of 16 positions; 64 tiles per block (8 waves x 8 tiles)
  const int n_tiles = (GB_B * GB_N * GB_N) / 16;                     // 32768
  const int blocks  = n_tiles / (WAVES_PER_BLOCK * TILES_PER_WAVE);  // 512
  gb_interior_kernel<<<blocks, WAVES_PER_BLOCK * 32, 0, stream>>>(
      ab, spm, eidx, Ew, Wd, Sw, out);

  // boundary: 32*32*257 elements
  const int btot = GB_B * GB_H * (GB_NP1 + GB_N);
  gb_boundary_kernel<<<(btot + 255) / 256, 256, 0, stream>>>(ab, gt, out);
}
